// CorrectedDH_SNN_86474871538189
// MI455X (gfx1250) — matrix operand order repaired and checked by reading, not compiled, not run on hardware
//
#include <hip/hip_runtime.h>
#include <hip/hip_bf16.h>

// CDNA5 (gfx1250) fused dendritic-SNN, round 3 (final polish).
//  - bf16 WMMA (v_wmma_f32_16x16x32_bf16) for the recurrent GEMM
//  - Tensor Data Mover (tensor_load_to_lds, TENSORcnt) streams the
//    L2-resident W slice into LDS, freeing VMEM/VALU for the A-tile cvt
//  - K split into two 512 phases (pure-x / pure-spike, template-constant),
//    K-chunk 64, double-buffered A in LDS, fully unrolled so the scheduler
//    software-pipelines global loads across iterations
//  - global_prefetch of x_{t+1} overlaps the sole HBM stream with compute

typedef __attribute__((ext_vector_type(16))) __bf16 v16bf;
typedef __attribute__((ext_vector_type(8)))  float  v8f;
typedef __attribute__((ext_vector_type(4)))  unsigned int v4u;
typedef __attribute__((ext_vector_type(8)))  int v8i;
typedef __attribute__((ext_vector_type(4)))  int v4i;

#ifdef __has_builtin
#if __has_builtin(__builtin_amdgcn_tensor_load_to_lds)
#define HAS_TDM 1
#endif
#if __has_builtin(__builtin_amdgcn_s_wait_tensorcnt)
#define HAS_TWAIT 1
#endif
#endif
#ifndef HAS_TDM
#define HAS_TDM 0
#endif
#ifndef HAS_TWAIT
#define HAS_TWAIT 0
#endif

namespace {
constexpr int Bb   = 128;   // batch
constexpr int Tt   = 1024;  // timesteps
constexpr int INn  = 512;   // input dim
constexpr int Hh   = 512;   // hidden
constexpr int OUTo = 128;   // readout
constexpr int Dd   = 1024;  // IN + H
constexpr int Nn   = 2048;  // H * NB
}

union Frag16 { v16bf v; uint4 q[2]; };

// ---------------- setup kernels ----------------

__global__ void k_cvt_bf16(const float* __restrict__ src,
                           __bf16* __restrict__ dst, int n) {
  int i = blockIdx.x * blockDim.x + threadIdx.x;
  if (i < n) dst[i] = (__bf16)src[i];
}

__global__ void k_params(const float* __restrict__ tau_n,
                         const float* __restrict__ tau_m,
                         float* __restrict__ beta,
                         float* __restrict__ alpha) {
  int i = blockIdx.x * blockDim.x + threadIdx.x;
  if (i < Nn) beta[i]  = 1.0f / (1.0f + expf(-tau_n[i]));  // [H,NB] flat == n
  if (i < Hh) alpha[i] = 1.0f / (1.0f + expf(-tau_m[i]));
}

__global__ void k_init(const float* __restrict__ mem0,
                       const float* __restrict__ spike0,
                       float* __restrict__ mem,
                       __bf16* __restrict__ spike,
                       float* __restrict__ dstate) {
  int i = blockIdx.x * blockDim.x + threadIdx.x;
  if (i < Bb * Hh) { mem[i] = mem0[i]; spike[i] = (__bf16)spike0[i]; }
  if (i < Bb * Nn) dstate[i] = 0.0f;
}

// ---------------- TDM helper ----------------
// One 2-D tensor DMA: rows x cols bf16 tile, row stride in elements, packed
// contiguously at lds_dst. D# bitfields per CDNA5 ISA ch.8.

#if HAS_TDM
__device__ __forceinline__ void tdm_load_tile(const __bf16* gsrc, void* lds_dst,
                                              int rows, int cols,
                                              int row_stride_elems) {
  unsigned long long ga = (unsigned long long)(size_t)gsrc;
  unsigned lds = (unsigned)(size_t)lds_dst;  // LDS aperture: addr[31:0] = offset
  v4u g0;
  g0.x = 1u;                                   // count=1, user descriptor
  g0.y = lds;                                  // lds_addr
  g0.z = (unsigned)(ga & 0xFFFFFFFFu);         // global_addr[31:0]
  g0.w = (unsigned)((ga >> 32) & 0x01FFFFFFu)  // global_addr[56:32]
         | (2u << 30);                         // type = 2 (image/tensor)
  v8i g1;
  g1[0] = (1 << 16);                           // data_size=1 -> 2B, no multicast
  g1[1] = (cols & 0xFFFF) << 16;               // tensor_dim0[15:0]
  g1[2] = ((unsigned)cols >> 16) | ((rows & 0xFFFF) << 16);  // dim0 hi | dim1 lo
  g1[3] = (cols & 0xFFFF) << 16;               // dim1 hi=0 | tile_dim0
  g1[4] = rows & 0xFFFF;                       // tile_dim1 | tile_dim2=0
  g1[5] = row_stride_elems;                    // tensor_dim0_stride[31:0]
  g1[6] = 0;                                   // stride0 hi | stride1 lo
  g1[7] = 0;                                   // stride1 hi
  v4i z4 = {0, 0, 0, 0};
#if __clang_major__ >= 23
  v8i z8 = {0, 0, 0, 0, 0, 0, 0, 0};
  __builtin_amdgcn_tensor_load_to_lds(g0, g1, z4, z4, z8, 0);
#else
  __builtin_amdgcn_tensor_load_to_lds(g0, g1, z4, z4, 0);
#endif
}

__device__ __forceinline__ void tdm_wait0() {
#if HAS_TWAIT
  __builtin_amdgcn_s_wait_tensorcnt(0);
#else
  asm volatile("s_wait_tensorcnt 0x0" ::: "memory");
#endif
}
#endif  // HAS_TDM

// ---------------- A-tile staging (32 contiguous cols of one row) ----------

template <bool ISX>
__device__ __forceinline__ void stage_a(const float* __restrict__ xrow,
                                        const __bf16* __restrict__ srow,
                                        __bf16* __restrict__ dst, int off) {
  __bf16 tmp[32];
  if (ISX) {  // input_spike: f32 -> bf16
    const float4* s4 = (const float4*)(xrow + off);
#pragma unroll
    for (int ii = 0; ii < 8; ++ii) {
      float4 f = s4[ii];
      tmp[4 * ii + 0] = (__bf16)f.x; tmp[4 * ii + 1] = (__bf16)f.y;
      tmp[4 * ii + 2] = (__bf16)f.z; tmp[4 * ii + 3] = (__bf16)f.w;
    }
  } else {    // previous spikes: already bf16
    const uint4* s4 = (const uint4*)(srow + off);
#pragma unroll
    for (int ii = 0; ii < 4; ++ii) ((uint4*)tmp)[ii] = s4[ii];
  }
  uint4* d = (uint4*)dst;
#pragma unroll
  for (int ii = 0; ii < 4; ++ii) d[ii] = ((uint4*)tmp)[ii];
}

// ---------------- one 512-wide K phase of the step GEMM ------------------

template <bool ISX>
__device__ __forceinline__ void gemm_phase(
    const float* __restrict__ xrow, const __bf16* __restrict__ srow,
    const __bf16* __restrict__ wsrc,
    __bf16* __restrict__ a_sh, __bf16* __restrict__ b_sh,
    int tid, int wave, int arow, int ac0,
    int mrow, int akof, int bkof, int ncl, v8f* acc) {
#if HAS_TDM
  if (wave == 0) {                 // one DMA brings the whole 32x512 W slice
    tdm_load_tile(wsrc, b_sh, 32, 512, Dd);
    tdm_wait0();
  }
#else
  for (int idx = tid * 8; idx < 32 * 512; idx += 256 * 8) {
    const int r = idx >> 9, c = idx & 511;
    *(uint4*)&b_sh[idx] = *(const uint4*)&wsrc[(size_t)r * Dd + c];
  }
#endif
  stage_a<ISX>(xrow, srow, &a_sh[arow * 64 + ac0], ac0);
  __syncthreads();  // B slice + A chunk 0 visible

#pragma unroll
  for (int it = 0; it < 8; ++it) {
    if (it + 1 < 8)  // prefetch next A chunk into the other buffer
      stage_a<ISX>(xrow, srow,
                   &a_sh[((it + 1) & 1) * (128 * 64) + arow * 64 + ac0],
                   (it + 1) * 64 + ac0);
    const __bf16* abuf = a_sh + (it & 1) * (128 * 64);
#pragma unroll
    for (int s = 0; s < 2; ++s) {
      Frag16 af;
      const __bf16* ap = &abuf[mrow * 64 + s * 32 + akof];
      af.q[0] = ((const uint4*)ap)[0];
      af.q[1] = *(const uint4*)(ap + 16);
#pragma unroll
      for (int j = 0; j < 2; ++j) {
        Frag16 bf;
        const __bf16* bp = &b_sh[(j * 16 + ncl) * 512 + it * 64 + s * 32 + bkof];
        bf.q[0] = ((const uint4*)bp)[0];
        bf.q[1] = *(const uint4*)(bp + 16);
        acc[j] = __builtin_amdgcn_wmma_f32_16x16x32_bf16(
            false, af.v, false, bf.v, (short)0, acc[j], false, false);
      }
    }
    __syncthreads();
  }
}

// ---------------- one recurrent timestep ----------------
// grid = N/32 = 64 WGs, 256 threads (8 waves). Wave w owns rows [16w,16w+16)
// x cols [n0,n0+32).

__global__ __launch_bounds__(256)
void snn_step(const float* __restrict__ x, int t,
              const __bf16* __restrict__ Wb,
              const float* __restrict__ bias,
              const float* __restrict__ beta,
              const float* __restrict__ alpha,
              float* __restrict__ mem,
              float* __restrict__ dstate,
              const __bf16* __restrict__ spike_in,
              __bf16* __restrict__ spike_out) {
  __shared__ __bf16 a_sh[2][128 * 64];  // 2 x 16KB ping-pong
  __shared__ __bf16 b_sh[32 * 512];     // 32KB W slice for current phase

  const int tid  = threadIdx.x;
  const int wave = tid >> 5;
  const int lane = tid & 31;
  const int n0   = blockIdx.x * 32;

  const int arow = tid >> 1;           // A staging: 32 cols of one row
  const int ac0  = (tid & 1) << 5;

  const int mrow = (wave << 4) + (lane & 15);
  const int akof = (lane >> 4) << 3;   // A K-chunk base: 0 or 8
  const int bkof = (lane >> 4) << 3;   // B K-chunk base: 0 or 8
  const int ncl  = lane & 15;

  const float* xrow = x + (size_t)arow * (Tt * INn) + (size_t)t * INn;
  const __bf16* srow = spike_in + arow * Hh;

  v8f acc[2] = {};

  gemm_phase<true>(xrow, srow, Wb + (size_t)n0 * Dd, &a_sh[0][0], b_sh,
                   tid, wave, arow, ac0, mrow, akof, bkof, ncl, acc);
  gemm_phase<false>(xrow, srow, Wb + (size_t)n0 * Dd + 512, &a_sh[0][0], b_sh,
                    tid, wave, arow, ac0, mrow, akof, bkof, ncl, acc);

  // overlap next step's only HBM stream with this step's tail
  if (t + 1 < Tt) __builtin_prefetch(xrow + INn, 0, 1);

  // ---- fused epilogue: dendritic filter + branch sum + soma + heaviside ----
  const int mbase = (wave << 4) + ((lane >> 4) << 3);
#pragma unroll
  for (int j = 0; j < 2; ++j) {
    const int n    = n0 + j * 16 + ncl;
    const float bt = beta[n];
    const float bs = bias[n];
#pragma unroll
    for (int r = 0; r < 8; ++r) {
      const int m       = mbase + r;
      const float dense = acc[j][r] + bs;
      float* dp         = dstate + (size_t)m * Nn + n;
      const float dn    = bt * (*dp) + (1.0f - bt) * dense;
      *dp = dn;
      // 4 dendritic branches of one neuron sit in 4 adjacent lanes
      const float l = dn + __shfl_xor(dn, 1) + __shfl_xor(dn, 2);
      if ((lane & 3) == 0) {
        const int h    = n >> 2;
        const float al = alpha[h];
        const float sp = (float)spike_in[m * Hh + h];
        const float mm = mem[m * Hh + h] * al + (1.0f - al) * l - sp;  // v_th=1
        mem[m * Hh + h]       = mm;
        spike_out[m * Hh + h] = (__bf16)((mm - 1.0f > 0.0f) ? 1.0f : 0.0f);
      }
    }
  }
}

// ---------------- readout: spike_final @ W_out.T + b_out ----------------

__global__ __launch_bounds__(256)
void snn_readout(const __bf16* __restrict__ spike,
                 const __bf16* __restrict__ Wob,
                 const float* __restrict__ b_out,
                 float* __restrict__ out) {
  __shared__ __bf16 a_sh[128 * 32];
  __shared__ __bf16 b_sh[64 * 32];
  const int tid = threadIdx.x;
  const int wave = tid >> 5, lane = tid & 31;
  const int n0 = blockIdx.x * 64;
  const int arow = tid >> 1, ac0 = (tid & 1) << 4;
  const int bnl = tid >> 2, bc0 = (tid & 3) << 3;
  const int mrow = (wave << 4) + (lane & 15);
  const int akof = (lane >> 4) << 3, bkof = (lane >> 4) << 3, ncl = lane & 15;

  v8f acc[4] = {};
  for (int kk = 0; kk < Hh; kk += 32) {
    {
      const uint4* s = (const uint4*)(spike + arow * Hh + kk + ac0);
      uint4* d = (uint4*)&a_sh[arow * 32 + ac0];
      d[0] = s[0]; d[1] = s[1];
    }
    *(uint4*)&b_sh[bnl * 32 + bc0] =
        *(const uint4*)&Wob[(size_t)(n0 + bnl) * Hh + kk + bc0];
    __syncthreads();
    Frag16 af;
    af.q[0] = *(const uint4*)&a_sh[mrow * 32 + akof];
    af.q[1] = *(const uint4*)&a_sh[mrow * 32 + akof + 16];
#pragma unroll
    for (int j = 0; j < 4; ++j) {
      Frag16 bf;
      const __bf16* bp = &b_sh[(j * 16 + ncl) * 32 + bkof];
      bf.q[0] = ((const uint4*)bp)[0];
      bf.q[1] = *(const uint4*)(bp + 16);
      acc[j] = __builtin_amdgcn_wmma_f32_16x16x32_bf16(
          false, af.v, false, bf.v, (short)0, acc[j], false, false);
    }
    __syncthreads();
  }
  const int mbase = (wave << 4) + ((lane >> 4) << 3);
#pragma unroll
  for (int j = 0; j < 4; ++j) {
    const int n = n0 + j * 16 + ncl;
    const float bo = b_out[n];
#pragma unroll
    for (int r = 0; r < 8; ++r) out[(mbase + r) * OUTo + n] = acc[j][r] + bo;
  }
}

// ---------------- host side ----------------

extern "C" void kernel_launch(void* const* d_in, const int* in_sizes, int n_in,
                              void* d_out, int out_size, void* d_ws, size_t ws_size,
                              hipStream_t stream) {
  (void)in_sizes; (void)n_in; (void)out_size; (void)ws_size;
  const float* x      = (const float*)d_in[0];  // [B,T,IN]
  const float* W      = (const float*)d_in[1];  // [H*NB, D]
  const float* bias   = (const float*)d_in[2];  // [H*NB]
  const float* tau_m  = (const float*)d_in[3];  // [H]
  const float* tau_n  = (const float*)d_in[4];  // [H,NB]
  const float* W_out  = (const float*)d_in[5];  // [OUT,H]
  const float* b_out  = (const float*)d_in[6];  // [OUT]
  const float* mem0   = (const float*)d_in[7];  // [B,H]
  const float* spike0 = (const float*)d_in[8];  // [B,H]
  float* out = (float*)d_out;                   // [B,OUT]

  char* ws = (char*)d_ws;
  size_t off = 0;
  auto carve = [&](size_t bytes) -> void* {
    void* p = ws + off;
    off = (off + bytes + 255) & ~(size_t)255;
    return p;
  };
  __bf16* Wb     = (__bf16*)carve((size_t)Nn * Dd * 2);   // 4 MB, L2-resident
  __bf16* Wob    = (__bf16*)carve((size_t)OUTo * Hh * 2); // 128 KB
  float*  beta   = (float*)carve((size_t)Nn * 4);
  float*  alpha  = (float*)carve((size_t)Hh * 4);
  float*  mem    = (float*)carve((size_t)Bb * Hh * 4);
  float*  dstate = (float*)carve((size_t)Bb * Nn * 4);    // 1 MB
  __bf16* sp0    = (__bf16*)carve((size_t)Bb * Hh * 2);
  __bf16* sp1    = (__bf16*)carve((size_t)Bb * Hh * 2);

  k_cvt_bf16<<<(Nn * Dd + 255) / 256, 256, 0, stream>>>(W, Wb, Nn * Dd);
  k_cvt_bf16<<<(OUTo * Hh + 255) / 256, 256, 0, stream>>>(W_out, Wob, OUTo * Hh);
  k_params<<<(Nn + 255) / 256, 256, 0, stream>>>(tau_n, tau_m, beta, alpha);
  k_init<<<(Bb * Nn + 255) / 256, 256, 0, stream>>>(mem0, spike0, mem, sp0, dstate);

  __bf16* bufs[2] = {sp0, sp1};
  for (int t = 0; t < Tt; ++t) {
    snn_step<<<Nn / 32, 256, 0, stream>>>(x, t, Wb, bias, beta, alpha, mem,
                                          dstate, bufs[t & 1], bufs[(t + 1) & 1]);
  }
  // T even -> final spikes back in bufs[0]
  snn_readout<<<OUTo / 64, 256, 0, stream>>>(bufs[0], Wob, b_out, out);
}
